// temporalfuse_6485400617414
// MI455X (gfx1250) — matrix operand order repaired and checked by reading, not compiled, not run on hardware
//
#include <hip/hip_runtime.h>
#include <hip/hip_bf16.h>

typedef __bf16 v16bf __attribute__((ext_vector_type(16)));
typedef float  v8f   __attribute__((ext_vector_type(8)));

#define T_    16
#define DIM_  256
#define HW_   128
#define HD_   64
#define SCALE_ 0.125f
#define PITCH 264   // 256 + 8 shorts pad: row stride 528B -> 4-bank lane stride, conflict-free b128 reads

static __device__ __forceinline__ unsigned short f2bfu(float x) {
    union { __bf16 h; unsigned short u; } c;
    c.h = (__bf16)x;
    return c.u;
}

// Blocks [0,256): pack Wo -> bf16 in WMMA A-fragment lane layout.
// Block 256: Q/K projections + softmax diagonal -> diag_out[4*16].
__global__ void tf_prep_kernel(const float* __restrict__ x,  const float* __restrict__ y,
                               const float* __restrict__ Wq, const float* __restrict__ bq,
                               const float* __restrict__ Wk, const float* __restrict__ bk,
                               const float* __restrict__ Wo,
                               float* __restrict__ diag_out, unsigned short* __restrict__ WoA) {
    const int tid = threadIdx.x;

    if (blockIdx.x < 256) {
        // A-fragment packing used here (contiguous k order, same as the B reads): lane l holds row m=l&15,
        // half j (0..15) -> K = (l>>4)*16 + j
        int gid   = blockIdx.x * 256 + tid;          // [0, 65536)
        int j     = gid & 15;
        int lane  = (gid >> 4) & 31;
        int mtile = (gid >> 9) & 15;
        int ktile = gid >> 13;                       // [0,8)
        int K = ktile * 32 + ((lane >> 4) << 4) + j;
        int o = mtile * 16 + (lane & 15);
        WoA[gid] = f2bfu(Wo[o * DIM_ + K]);
        return;
    }

    __shared__ float QL[T_][DIM_];
    __shared__ float KL[T_][DIM_];

    const int c = tid;                               // output channel of projection
    float accQ[T_], accK[T_];
#pragma unroll
    for (int t = 0; t < T_; ++t) { accQ[t] = bq[c]; accK[t] = bk[c]; }
    for (int d = 0; d < DIM_; ++d) {
        float wq = Wq[c * DIM_ + d];
        float wk = Wk[c * DIM_ + d];
#pragma unroll
        for (int t = 0; t < T_; ++t) {
            accQ[t] = fmaf(x[t * DIM_ + d], wq, accQ[t]);
            accK[t] = fmaf(y[t * DIM_ + d], wk, accK[t]);
        }
    }
#pragma unroll
    for (int t = 0; t < T_; ++t) { QL[t][c] = accQ[t]; KL[t][c] = accK[t]; }
    __syncthreads();

    if (tid < 64) {
        int h = tid >> 4, t = tid & 15;
        float s[T_];
#pragma unroll
        for (int k = 0; k < T_; ++k) {
            float a = 0.f;
            for (int d = 0; d < HD_; ++d)
                a = fmaf(QL[t][h * HD_ + d], KL[k][h * HD_ + d], a);
            s[k] = a * SCALE_;
        }
        float m = s[0];
#pragma unroll
        for (int k = 1; k < T_; ++k) m = fmaxf(m, s[k]);
        float den = 0.f;
#pragma unroll
        for (int k = 0; k < T_; ++k) den += __expf(s[k] - m);
        diag_out[tid] = __expf(s[t] - m) / den;      // tid == h*16 + t
    }
}

// Main GEMM: out[t,o,hh,ww] = z[t,o,hh,ww] + bo[o] + diag[h,t] * sum_c Wo[o,c]*v[c,p]
// Workgroup: fixed (t, h=hh%4, wwblock); pixel tile = 64 (all 32 q=hh/4, 2 ww).
// Pixel map: p = 2*q + wwi  (ww fastest -> adjacent lanes store adjacent addresses).
__global__ void __launch_bounds__(128)
tf_main_kernel(const float* __restrict__ z, const float* __restrict__ bo,
               const float* __restrict__ diag, const unsigned short* __restrict__ WoA,
               float* __restrict__ out) {
    __shared__ unsigned short Vlds[64 * PITCH];      // 64 pixels x 256 c, bf16

    const int wwblock = blockIdx.x;                  // 0..63 (2 ww each)
    const int h       = blockIdx.y;                  // 0..3
    const int t       = blockIdx.z;                  // 0..15
    const int tid     = threadIdx.x;                 // 0..127

    const float* zt = z + (size_t)t * DIM_ * HW_ * HW_;

    // ---- Gather permuted z tile -> LDS (coalesced float4 along full W rows) ----
    // needed: cz in [64h,64h+64), H in {2 ww}, all 128 W
    for (int it = 0; it < 32; ++it) {
        int f    = tid + 128 * it;                   // [0, 4096) float4 segments
        int wseg = f & 31;                           // W segment of 4
        int wwi  = (f >> 5) & 1;
        int ci   = f >> 6;                           // [0,64)
        int cz   = h * 64 + ci;
        int ww   = wwblock * 2 + wwi;
        int Wb   = wseg * 4;
        const float4 v = *reinterpret_cast<const float4*>(zt + ((size_t)cz * HW_ + ww) * HW_ + Wb);
        int cc = ci * 4 + (Wb >> 5);                 // c = 4*(c/4) + W/32  (same for all 4)
        int q0 = Wb & 31;                            // q = W % 32
        // p = 2*q + wwi
        Vlds[(2 * (q0 + 0) + wwi) * PITCH + cc] = f2bfu(v.x);
        Vlds[(2 * (q0 + 1) + wwi) * PITCH + cc] = f2bfu(v.y);
        Vlds[(2 * (q0 + 2) + wwi) * PITCH + cc] = f2bfu(v.z);
        Vlds[(2 * (q0 + 3) + wwi) * PITCH + cc] = f2bfu(v.w);
    }
    __syncthreads();

    // ---- WMMA GEMM: A = Wo tile (16o x 32c), B = V tile (32c x 16p) ----
    const int lane  = tid & 31;
    const int wave  = tid >> 5;                      // N-tile id (0..3)
    const int pfrag = wave * 16 + (lane & 15);       // B-frag: lane column = pixel
    const int khalf = lane >> 4;                     // B-frag: K = khalf*16 + j

    // Load all 8 B-fragments once (64 VGPRs, LDS-resident tile).
    union BF { v16bf v; uint4 u[2]; };
    BF bfr[8];
#pragma unroll
    for (int kt = 0; kt < 8; ++kt) {
        const uint4* bp = reinterpret_cast<const uint4*>(
            &Vlds[pfrag * PITCH + kt * 32 + khalf * 16]);
        bfr[kt].u[0] = bp[0];
        bfr[kt].u[1] = bp[1];
    }

    // Epilogue geometry (per-lane constants)
    const float dv  = diag[h * 16 + t];
    const int n     = lane & 15;
    const int p     = wave * 16 + n;
    const int wwi2  = p & 1;
    const int q     = p >> 1;
    const int hh    = q * 4 + h;
    const int ww2   = wwblock * 2 + wwi2;
    const int mhi   = (lane >> 4) * 8;               // D-layout: m = r + 8*(lane>=16)
    const size_t pixoff = ((size_t)t * DIM_ * HW_ + hh) * HW_ + ww2;
    const float* zres = z + pixoff;
    float* op = out + pixoff;
    const float* bop = bo + mhi;                     // per-lane contiguous bias rows

    v8f zero8 = {0.f, 0.f, 0.f, 0.f, 0.f, 0.f, 0.f, 0.f};

    // m-tile outer loop: single 8-VGPR accumulator; 16 clause-loaded A b128s
    // cover 8 back-to-back WMMAs; batched residual/bias loads (one wait) then
    // a clean fma+store sweep, all overlappable with the next tile's loads.
#pragma unroll
    for (int mt = 0; mt < 16; ++mt) {
        union AF { v16bf v; uint4 u[2]; };
        AF afr[8];
#pragma unroll
        for (int kt = 0; kt < 8; ++kt) {
            const uint4* ap = reinterpret_cast<const uint4*>(
                WoA + ((((kt * 16 + mt) * 32) + lane) << 4));
            afr[kt].u[0] = ap[0];
            afr[kt].u[1] = ap[1];
        }

        // Batch the epilogue input loads (independent of acc -> schedulable early).
        float res[8];
#pragma unroll
        for (int r = 0; r < 8; ++r)
            res[r] = zres[(size_t)(mt * 16 + mhi + r) * (HW_ * HW_)];
        const float4 b0 = *reinterpret_cast<const float4*>(bop + mt * 16);
        const float4 b1 = *reinterpret_cast<const float4*>(bop + mt * 16 + 4);

        v8f acc = zero8;
#pragma unroll
        for (int kt = 0; kt < 8; ++kt) {
            acc = __builtin_amdgcn_wmma_f32_16x16x32_bf16(
                false, afr[kt].v, false, bfr[kt].v, (short)0, acc, false, false);
        }

        float bias[8] = {b0.x, b0.y, b0.z, b0.w, b1.x, b1.y, b1.z, b1.w};
#pragma unroll
        for (int r = 0; r < 8; ++r) {
            size_t off = (size_t)(mt * 16 + mhi + r) * (HW_ * HW_);
            op[off] = res[r] + bias[r] + dv * acc[r];
        }
    }
}

extern "C" void kernel_launch(void* const* d_in, const int* in_sizes, int n_in,
                              void* d_out, int out_size, void* d_ws, size_t ws_size,
                              hipStream_t stream) {
    (void)in_sizes; (void)n_in; (void)out_size; (void)ws_size;
    const float* x  = (const float*)d_in[0];
    const float* y  = (const float*)d_in[1];
    const float* z  = (const float*)d_in[2];
    const float* Wq = (const float*)d_in[3];
    const float* bq = (const float*)d_in[4];
    const float* Wk = (const float*)d_in[5];
    const float* bk = (const float*)d_in[6];
    const float* Wo = (const float*)d_in[7];
    const float* bo = (const float*)d_in[8];
    float* out = (float*)d_out;

    float* diag = (float*)d_ws;                                   // 64 floats
    unsigned short* WoA = (unsigned short*)((char*)d_ws + 512);   // 65536 bf16 (128KB)

    tf_prep_kernel<<<dim3(257), dim3(256), 0, stream>>>(x, y, Wq, bq, Wk, bk, Wo, diag, WoA);
    tf_main_kernel<<<dim3(64, 4, 16), dim3(128), 0, stream>>>(z, bo, diag, WoA, out);
}
